// SwinLikePyramidTM_13013750907236
// MI455X (gfx1250) — compile-verified, matrix operand order checked by reading, not run on hardware
//
#include <hip/hip_runtime.h>

// ---------------------------------------------------------------------------
// MI455X (gfx1250, wave32) implementation.
// Memory-bound problem (~4.5 GFLOP vs ~100MB HBM @ 23.3 TB/s) -> single pass
// per tensor, bf16 WMMA (v_wmma_f32_16x16x32_bf16) for all three GEMMs, the
// fuzzy-TM min-reduction replaced by 512-bit mask logic, and GEMM tile
// staging with double-buffered GLOBAL_LOAD_ASYNC_TO_LDS_B128: the next tile's
// global traffic is in flight (ASYNCcnt=8) while the current tile feeds WMMA;
// in-order async completion lets s_wait_asynccnt 0x4 release exactly the
// current tile.
// ---------------------------------------------------------------------------

typedef __attribute__((ext_vector_type(16))) __bf16        v16bf;
typedef __attribute__((ext_vector_type(8)))  float         v8f;
typedef __attribute__((ext_vector_type(8)))  unsigned int  v8u;
typedef __attribute__((ext_vector_type(4)))  unsigned int  v4u;

#define NBATCH 8
#define HDIM   56
#define WDIM   56
#define CDIM   192
#define SHIFT2 2
#define NWIN   1568   // 8 * 14 * 14
#define IND    3072   // 4*4*192
#define HID    256
#define NCL    256
#define TILE_E (64 * 32)   // one 64x32 bf16 tile in LDS

__device__ __forceinline__ unsigned short f2bf(float f) {
  unsigned u = __builtin_bit_cast(unsigned, f);
  u = (u + 0x7FFFu + ((u >> 16) & 1u)) >> 16;   // round-to-nearest-even
  return (unsigned short)u;
}
__device__ __forceinline__ float sigm(float x) { return 1.0f / (1.0f + __expf(-x)); }

// ---------------------------------------------------------------------------
// Kernel 1: LayerNorm + roll(-2,-2) + 4x4 window partition -> bf16
// ---------------------------------------------------------------------------
__global__ void ln_roll_part(const float* __restrict__ x,
                             const float* __restrict__ gamma,
                             const float* __restrict__ beta,
                             unsigned short* __restrict__ winb) {
  __shared__ float ssum[192];
  __shared__ float ssq[192];
  __shared__ float mv[2];
  int t  = threadIdx.x;
  int sp = blockIdx.x;                       // 0 .. 25087
  int w  = sp % WDIM;
  int h  = (sp / WDIM) % HDIM;
  int ib = sp / (WDIM * HDIM);
  float v = x[(size_t)sp * CDIM + t];
  ssum[t] = v; ssq[t] = v * v;
  __syncthreads();
  for (int s = 96; s >= 3; s >>= 1) {        // 96,48,24,12,6,3
    if (t < s) { ssum[t] += ssum[t + s]; ssq[t] += ssq[t + s]; }
    __syncthreads();
  }
  if (t == 0) {
    float m = (ssum[0] + ssum[1] + ssum[2]) * (1.0f / 192.0f);
    float q = (ssq[0]  + ssq[1]  + ssq[2])  * (1.0f / 192.0f);
    mv[0] = m; mv[1] = q - m * m;
  }
  __syncthreads();
  float y = (v - mv[0]) * rsqrtf(mv[1] + 1e-5f) * gamma[t] + beta[t];
  int h2 = (h - SHIFT2 + HDIM) % HDIM;       // position after roll(-2)
  int w2 = (w - SHIFT2 + WDIM) % WDIM;
  int bwin = (ib * 14 + (h2 >> 2)) * 14 + (w2 >> 2);
  int j    = ((((h2 & 3) << 2) | (w2 & 3)) * CDIM) + t;
  winb[(size_t)bwin * IND + j] = f2bf(y);
}

// ---------------------------------------------------------------------------
// Kernel 2: transpose + f32->bf16 pack of a (K x N) weight into (N x K)
// ---------------------------------------------------------------------------
__global__ void tpack(const float* __restrict__ src, unsigned short* __restrict__ dst,
                      int K, int N) {
  int i = blockIdx.x * blockDim.x + threadIdx.x;
  if (i >= K * N) return;
  int k = i / N, n = i % N;
  dst[(size_t)n * K + k] = f2bf(src[i]);
}

// ---------------------------------------------------------------------------
// WMMA GEMM core: block = 128 threads (4 waves), 64M x 64N tile, K step 32,
// double-buffered async LDS staging. A is (M x K) bf16 row-major; BT is
// (N x K) bf16 row-major. Row indices are clamped instead of zero-filled:
// out-of-range rows only feed accumulator rows whose stores are guarded,
// and N is an exact multiple of 64 in all three GEMMs.
// ---------------------------------------------------------------------------
__device__ __forceinline__ void stage_async(const unsigned short* __restrict__ G,
                                            int rows, int ld, int rbase, int k0,
                                            unsigned short* __restrict__ sh, int tid) {
  // 64 rows x 32 cols of bf16 = 256 x 16B chunks, 128 threads -> 2 each
#pragma unroll
  for (int i = 0; i < 2; ++i) {
    int c = tid + i * 128;
    int r = c >> 2, off = (c & 3) << 3;
    int gr = rbase + r;
    gr = gr < rows ? gr : rows - 1;                       // clamp (no branch)
    const unsigned short* gp = G + (size_t)gr * ld + k0 + off;
    unsigned lds = (unsigned)(uintptr_t)(sh + r * 32 + off);  // low 32b = LDS offset
    asm volatile("global_load_async_to_lds_b128 %0, %1, off"
                 :: "v"(lds), "v"(gp) : "memory");
  }
}

__device__ __forceinline__ v16bf frag(const unsigned short* sh, int row, int kh) {
  v4u lo = *(const v4u*)(sh + row * 32 + kh);       // K = kh..kh+7
  v4u hi = *(const v4u*)(sh + row * 32 + kh + 16);  // K = kh+16..kh+23
  v8u u;
  u[0] = lo[0]; u[1] = lo[1]; u[2] = lo[2]; u[3] = lo[3];
  u[4] = hi[0]; u[5] = hi[1]; u[6] = hi[2]; u[7] = hi[3];
  return __builtin_bit_cast(v16bf, u);
}

__device__ __forceinline__ void gemm64(const unsigned short* __restrict__ A,
                                       const unsigned short* __restrict__ BT,
                                       int M, int N, int K, int mBase, int nBase,
                                       unsigned short* sA, unsigned short* sB,  // 2 tiles each
                                       v8f acc[4]) {
  int tid = threadIdx.x, wave = tid >> 5, lane = tid & 31;
  int lrow = lane & 15, kh = (lane >> 4) << 3;

  // prologue: tile 0 -> buffer 0   (4 async instructions per wave per tile)
  stage_async(A,  M, K, mBase, 0, sA, tid);
  stage_async(BT, N, K, nBase, 0, sB, tid);

  for (int k0 = 0; k0 < K; k0 += 32) {
    int cur = (k0 >> 5) & 1;
    unsigned short* cA = sA + cur * TILE_E;
    unsigned short* cB = sB + cur * TILE_E;
    if (k0 + 32 < K) {
      int nxt = cur ^ 1;
      stage_async(A,  M, K, mBase, k0 + 32, sA + nxt * TILE_E, tid);
      stage_async(BT, N, K, nBase, k0 + 32, sB + nxt * TILE_E, tid);
      // L2 warm-up for the tile after next
      if (k0 + 64 < K) {
        int pr = mBase + (tid & 63); pr = pr < M ? pr : M - 1;
        __builtin_prefetch(A + (size_t)pr * K + k0 + 64, 0, 1);
        int pn = nBase + (tid & 63); pn = pn < N ? pn : N - 1;
        __builtin_prefetch(BT + (size_t)pn * K + k0 + 64, 0, 1);
      }
      // 8 outstanding; in-order completion => <=4 means current tile landed
      asm volatile("s_wait_asynccnt 0x4" ::: "memory");
    } else {
      asm volatile("s_wait_asynccnt 0x0" ::: "memory");
    }
    __syncthreads();
    v16bf af = frag(cA, wave * 16 + lrow, kh);
#pragma unroll
    for (int nt = 0; nt < 4; ++nt) {
      v16bf bf = frag(cB, nt * 16 + lrow, kh);
      acc[nt] = __builtin_amdgcn_wmma_f32_16x16x32_bf16(false, af, false, bf,
                                                        (short)0, acc[nt],
                                                        false, false);
    }
    __syncthreads();   // all waves done reading `cur` before it is recycled
  }
}

// ---------------------------------------------------------------------------
// GEMM1: preact = win @ W1 + b1   (1568 x 256), stored f32
// ---------------------------------------------------------------------------
__global__ void gemm1_preact(const unsigned short* __restrict__ A,
                             const unsigned short* __restrict__ BT,
                             const float* __restrict__ b1,
                             float* __restrict__ preact) {
  __shared__ __align__(16) unsigned short sA[2 * TILE_E];
  __shared__ __align__(16) unsigned short sB[2 * TILE_E];
  v8f z = {0.f, 0.f, 0.f, 0.f, 0.f, 0.f, 0.f, 0.f};
  v8f acc[4] = {z, z, z, z};
  int mBase = blockIdx.x * 64, nBase = blockIdx.y * 64;
  gemm64(A, BT, NWIN, HID, IND, mBase, nBase, sA, sB, acc);
  int lane = threadIdx.x & 31, wave = threadIdx.x >> 5;
  int rowOff = wave * 16 + ((lane >> 4) << 3);
  int colOff = lane & 15;
#pragma unroll
  for (int nt = 0; nt < 4; ++nt)
#pragma unroll
    for (int r = 0; r < 8; ++r) {
      int grow = mBase + rowOff + r;
      int gcol = nBase + nt * 16 + colOff;
      if (grow < NWIN)
        preact[(size_t)grow * HID + gcol] = acc[nt][r] + b1[gcol];
    }
}

// ---------------------------------------------------------------------------
// bit-pack preact signs: bits[b][f] = (preact > 0)  (8 u32 per window)
// ---------------------------------------------------------------------------
__global__ void bitpack(const float* __restrict__ preact,
                        unsigned int* __restrict__ bits) {
  __shared__ unsigned int wsh[8];
  int t = threadIdx.x, b = blockIdx.x;
  if (t < 8) wsh[t] = 0u;
  __syncthreads();
  if (preact[(size_t)b * HID + t] > 0.0f) atomicOr(&wsh[t >> 5], 1u << (t & 31));
  __syncthreads();
  if (t < 8) bits[b * 8 + t] = wsh[t];
}

// ---------------------------------------------------------------------------
// clause include masks: P = (inc_w[:, :256] > 0), N = (inc_w[:, 256:] > 0)
// ---------------------------------------------------------------------------
__global__ void incmask(const float* __restrict__ incw,
                        unsigned int* __restrict__ P,
                        unsigned int* __restrict__ Nm) {
  __shared__ unsigned int ps[8], ns[8];
  int t = threadIdx.x, cl = blockIdx.x;
  if (t < 8) { ps[t] = 0u; ns[t] = 0u; }
  __syncthreads();
  if (incw[(size_t)cl * 512 + t] > 0.0f)        atomicOr(&ps[t >> 5], 1u << (t & 31));
  if (incw[(size_t)cl * 512 + 256 + t] > 0.0f)  atomicOr(&ns[t >> 5], 1u << (t & 31));
  __syncthreads();
  if (t < 8) { P[cl * 8 + t] = ps[t]; Nm[cl * 8 + t] = ns[t]; }
}

// ---------------------------------------------------------------------------
// clauses: clause=1 iff every included literal > 0.5
//   fail = (P & ~bits) | (N & bits)  over 8 words
// ---------------------------------------------------------------------------
__global__ void clause_k(const unsigned int* __restrict__ bits,
                         const unsigned int* __restrict__ P,
                         const unsigned int* __restrict__ Nm,
                         float* __restrict__ clausesF,
                         unsigned short* __restrict__ clausesB,
                         float* __restrict__ summary) {
  __shared__ unsigned int wb[8];
  __shared__ float red[256];
  int t = threadIdx.x, b = blockIdx.x;   // t = clause index
  if (t < 8) wb[t] = bits[b * 8 + t];
  __syncthreads();
  unsigned fail = 0u;
#pragma unroll
  for (int w = 0; w < 8; ++w) {
    unsigned bw = wb[w];
    fail |= (P[t * 8 + w] & ~bw) | (Nm[t * 8 + w] & bw);
  }
  float cv = fail ? 0.0f : 1.0f;
  clausesF[(size_t)b * NCL + t] = cv;
  clausesB[(size_t)b * NCL + t] = f2bf(cv);
  red[t] = cv;
  __syncthreads();
  for (int s = 128; s > 0; s >>= 1) {
    if (t < s) red[t] += red[t + s];
    __syncthreads();
  }
  if (t == 0) summary[b] = red[0] * (1.0f / 256.0f);
}

// ---------------------------------------------------------------------------
// GEMM2: logits = clauses @ vote_w / 16  (1568 x 192), bf16 out
// ---------------------------------------------------------------------------
__global__ void gemm2_logits(const unsigned short* __restrict__ A,
                             const unsigned short* __restrict__ BT,
                             unsigned short* __restrict__ logitsb) {
  __shared__ __align__(16) unsigned short sA[2 * TILE_E];
  __shared__ __align__(16) unsigned short sB[2 * TILE_E];
  v8f z = {0.f, 0.f, 0.f, 0.f, 0.f, 0.f, 0.f, 0.f};
  v8f acc[4] = {z, z, z, z};
  int mBase = blockIdx.x * 64, nBase = blockIdx.y * 64;
  gemm64(A, BT, NWIN, CDIM, NCL, mBase, nBase, sA, sB, acc);
  int lane = threadIdx.x & 31, wave = threadIdx.x >> 5;
  int rowOff = wave * 16 + ((lane >> 4) << 3);
  int colOff = lane & 15;
#pragma unroll
  for (int nt = 0; nt < 4; ++nt)
#pragma unroll
    for (int r = 0; r < 8; ++r) {
      int grow = mBase + rowOff + r;
      int gcol = nBase + nt * 16 + colOff;
      if (grow < NWIN)
        logitsb[(size_t)grow * CDIM + gcol] = f2bf(acc[nt][r] * 0.0625f);
    }
}

// ---------------------------------------------------------------------------
// GEMM3: feat = sigmoid(logits @ W2 + b2), fused window-reverse + roll(+2,+2)
//        + gate blend + residual (bijective scatter -> one store per element)
// ---------------------------------------------------------------------------
__global__ void gemm3_out(const unsigned short* __restrict__ A,
                          const unsigned short* __restrict__ BT,
                          const float* __restrict__ b2,
                          const float* __restrict__ x,
                          const float* __restrict__ gate,
                          float* __restrict__ out) {
  __shared__ __align__(16) unsigned short sA[2 * TILE_E];
  __shared__ __align__(16) unsigned short sB[2 * TILE_E];
  v8f z = {0.f, 0.f, 0.f, 0.f, 0.f, 0.f, 0.f, 0.f};
  v8f acc[4] = {z, z, z, z};
  int mBase = blockIdx.x * 64, nBase = blockIdx.y * 64;
  gemm64(A, BT, NWIN, IND, CDIM, mBase, nBase, sA, sB, acc);
  int lane = threadIdx.x & 31, wave = threadIdx.x >> 5;
  int rowOff = wave * 16 + ((lane >> 4) << 3);
  int colOff = lane & 15;
  float g = sigm(gate[0]);
#pragma unroll
  for (int nt = 0; nt < 4; ++nt)
#pragma unroll
    for (int r = 0; r < 8; ++r) {
      int grow = mBase + rowOff + r;
      int gcol = nBase + nt * 16 + colOff;
      if (grow < NWIN) {
        float m = sigm(acc[nt][r] + b2[gcol]);
        int wy = gcol / (4 * CDIM);
        int wx = (gcol / CDIM) & 3;
        int c  = gcol % CDIM;
        int ib  = grow / 196;
        int rem = grow % 196;
        int h2 = (rem / 14) * 4 + wy;
        int w2 = (rem % 14) * 4 + wx;
        int h = (h2 + SHIFT2) % HDIM;
        int w = (w2 + SHIFT2) % WDIM;
        size_t oi = ((size_t)(ib * HDIM + h) * WDIM + w) * CDIM + c;
        out[oi] = x[oi] + g * m + (1.0f - g) * sigm(m);
      }
    }
}

// ---------------------------------------------------------------------------
// launch
// ---------------------------------------------------------------------------
extern "C" void kernel_launch(void* const* d_in, const int* in_sizes, int n_in,
                              void* d_out, int out_size, void* d_ws, size_t ws_size,
                              hipStream_t stream) {
  (void)in_sizes; (void)n_in; (void)out_size; (void)ws_size;

  const float* x      = (const float*)d_in[0];
  const float* gamma  = (const float*)d_in[1];
  const float* beta   = (const float*)d_in[2];
  const float* W1     = (const float*)d_in[3];
  const float* b1     = (const float*)d_in[4];
  const float* inc_w  = (const float*)d_in[5];
  const float* vote_w = (const float*)d_in[6];
  const float* W2     = (const float*)d_in[7];
  const float* b2     = (const float*)d_in[8];
  const float* gate   = (const float*)d_in[9];

  float* out = (float*)d_out;
  float* clausesF = out + (size_t)NBATCH * HDIM * WDIM * CDIM;   // +4,816,896
  float* summary  = clausesF + (size_t)NWIN * NCL;               // +401,408

  // workspace layout (all offsets 256B aligned)
  char* ws = (char*)d_ws;
  size_t o = 0;
  auto carve = [&](size_t bytes) { size_t r = o; o += (bytes + 255) & ~(size_t)255; return r; };
  unsigned short* winb     = (unsigned short*)(ws + carve((size_t)NWIN * IND * 2));
  unsigned short* W1T      = (unsigned short*)(ws + carve((size_t)HID * IND * 2));
  unsigned short* voteT    = (unsigned short*)(ws + carve((size_t)CDIM * NCL * 2));
  unsigned short* W2T      = (unsigned short*)(ws + carve((size_t)IND * CDIM * 2));
  float*          preact   = (float*)        (ws + carve((size_t)NWIN * HID * 4));
  unsigned int*   bits     = (unsigned int*) (ws + carve((size_t)NWIN * 8 * 4));
  unsigned int*   Pm       = (unsigned int*) (ws + carve((size_t)NCL * 8 * 4));
  unsigned int*   Nmk      = (unsigned int*) (ws + carve((size_t)NCL * 8 * 4));
  unsigned short* clausesB = (unsigned short*)(ws + carve((size_t)NWIN * NCL * 2));
  unsigned short* logitsB  = (unsigned short*)(ws + carve((size_t)NWIN * CDIM * 2));

  // weight packs (bf16, transposed so WMMA B-fragments are contiguous)
  tpack<<<(IND * HID + 255) / 256, 256, 0, stream>>>(W1, W1T, IND, HID);
  tpack<<<(NCL * CDIM + 255) / 256, 256, 0, stream>>>(vote_w, voteT, NCL, CDIM);
  tpack<<<(CDIM * IND + 255) / 256, 256, 0, stream>>>(W2, W2T, CDIM, IND);

  // LN + roll + partition
  ln_roll_part<<<NBATCH * HDIM * WDIM, CDIM, 0, stream>>>(x, gamma, beta, winb);

  // clause include masks
  incmask<<<NCL, 256, 0, stream>>>(inc_w, Pm, Nmk);

  // GEMM1 (1568x256 = win @ W1)
  gemm1_preact<<<dim3((NWIN + 63) / 64, HID / 64), 128, 0, stream>>>(winb, W1T, b1, preact);

  // sign bits + clauses (bitwise fuzzy-TM) + summary
  bitpack<<<NWIN, 256, 0, stream>>>(preact, bits);
  clause_k<<<NWIN, 256, 0, stream>>>(bits, Pm, Nmk, clausesF, clausesB, summary);

  // GEMM2 (1568x192 = clauses @ vote_w / 16)
  gemm2_logits<<<dim3((NWIN + 63) / 64, CDIM / 64), 128, 0, stream>>>(clausesB, voteT, logitsB);

  // GEMM3 (1568x3072) + fused epilogue -> out
  gemm3_out<<<dim3((NWIN + 63) / 64, IND / 64), 128, 0, stream>>>(logitsB, W2T, b2, x, gate, out);
}
